// DynamicAttention4_40827959116537
// MI455X (gfx1250) — matrix operand order repaired and checked by reading, not compiled, not run on hardware
//
#include <hip/hip_runtime.h>
#include <hip/hip_bf16.h>
#include <math.h>

// ---------- types ----------
typedef unsigned short u16;
typedef __attribute__((ext_vector_type(16))) __bf16 v16bf;
typedef __attribute__((ext_vector_type(8)))  float  v8f;
typedef __attribute__((ext_vector_type(16))) u16    u16x16;
typedef __attribute__((ext_vector_type(8)))  u16    u16x8;
typedef __attribute__((ext_vector_type(4)))  u16    u16x4;
typedef __attribute__((ext_vector_type(4)))  int    i32x4;

// ---------- optional CDNA5 async global->LDS path ----------
#if defined(__has_builtin)
#if __has_builtin(__builtin_amdgcn_global_load_async_to_lds_b128) && \
    __has_builtin(__builtin_amdgcn_s_wait_asynccnt)
#define ATTN_ASYNC 1
#else
#define ATTN_ASYNC 0
#endif
#else
#define ATTN_ASYNC 0
#endif

// ---------- bf16 <-> f32 (RNE), manual so storage stays ushort ----------
__device__ __forceinline__ u16 f2bf(float f) {
  union { float f; unsigned u; } c; c.f = f;
  unsigned u = c.u;
  u += 0x7fffu + ((u >> 16) & 1u);
  return (u16)(u >> 16);
}
__device__ __forceinline__ float bf2f(u16 h) {
  union { unsigned u; float f; } c; c.u = ((unsigned)h) << 16;
  return c.f;
}
__device__ __forceinline__ float fast_tanh(float x) {
  float e = __expf(2.0f * x);
  return 1.0f - 2.0f / (e + 1.0f);
}

// ---------- GEMM tile config ----------
#define BM 128   // block rows
#define BN 64    // block cols
#define BK 32    // K step
#define BKP 40   // padded LDS row pitch (80B) -> conflict-free b128 access

// stage 8 contiguous elements into LDS, converting to bf16 when source is f32
__device__ __forceinline__ void stage8v(const float* g, u16* s) {
  const float4 f0 = *(const float4*)g;
  const float4 f1 = *(const float4*)(g + 4);
  u16x8 o;
  o[0]=f2bf(f0.x); o[1]=f2bf(f0.y); o[2]=f2bf(f0.z); o[3]=f2bf(f0.w);
  o[4]=f2bf(f1.x); o[5]=f2bf(f1.y); o[6]=f2bf(f1.z); o[7]=f2bf(f1.w);
  *(u16x8*)s = o;
}
__device__ __forceinline__ void stage8v(const u16* g, u16* s) {
#if ATTN_ASYNC
  __builtin_amdgcn_global_load_async_to_lds_b128(
      (__attribute__((address_space(1))) i32x4*)g,
      (__attribute__((address_space(3))) i32x4*)s, 0, 0);
#else
  *(u16x8*)s = *(const u16x8*)g;
#endif
}
__device__ __forceinline__ void async_wait_then_sync() {
#if ATTN_ASYNC
  __builtin_amdgcn_s_wait_asynccnt(0);
#endif
  __syncthreads();
}

__device__ __forceinline__ void read8f(const float* g, float* o) {
  const float4 f0 = *(const float4*)g; const float4 f1 = *(const float4*)(g + 4);
  o[0]=f0.x; o[1]=f0.y; o[2]=f0.z; o[3]=f0.w; o[4]=f1.x; o[5]=f1.y; o[6]=f1.z; o[7]=f1.w;
}
__device__ __forceinline__ void read8f(const u16* g, float* o) {
  u16x8 v = *(const u16x8*)g;
#pragma unroll
  for (int i = 0; i < 8; ++i) o[i] = bf2f(v[i]);
}

// A fragment: 16x32 bf16. lane<16: M=lane, K 0-7 / 16-23 ; lane>=16: M=lane-16, K 8-15 / 24-31
__device__ __forceinline__ v16bf frag_a(const u16* tile, int lane) {
  const u16* p = tile + (lane & 15) * BKP + ((lane >> 4) << 3);
  u16x8 lo = *(const u16x8*)p;
  u16x8 hi = *(const u16x8*)(p + 16);
  u16x16 r;
#pragma unroll
  for (int i = 0; i < 8; ++i) { r[i] = lo[i]; r[i + 8] = hi[i]; }
  return __builtin_bit_cast(v16bf, r);
}
// B fragment from K-transposed LDS tile Bt[n][k]: lane<16: N=lane, K 0-7 / 16-23; lane>=16: N=lane-16, K 8-15 / 24-31
__device__ __forceinline__ v16bf frag_b(const u16* tile, int lane) {
  const u16* p = tile + (lane & 15) * BKP + ((lane >> 4) << 3);
  u16x8 lo = *(const u16x8*)p;
  u16x8 hi = *(const u16x8*)(p + 16);
  u16x16 r;
#pragma unroll
  for (int i = 0; i < 8; ++i) { r[i] = lo[i]; r[i + 8] = hi[i]; }
  return __builtin_bit_cast(v16bf, r);
}

__device__ __forceinline__ void store_c(float* p, float v) { *p = v; }
__device__ __forceinline__ void store_c(u16*   p, float v) { *p = f2bf(v); }

// C(M,N) = act(scale * A(M,K) @ B + bias). BT=true: B given as (N,K) row-major.
// grid = (N/64, M/128, batch), block = 256 = 8 waves; wave grid 4(M)x2(N); each wave: 32x32
// (4 WMMAs / K-step / wave), double-buffered LDS, one barrier per K-step.
template <typename TA, typename TB, typename TC, bool BT, int ACT>
__global__ __launch_bounds__(256) void gemm_bias_act(
    const TA* __restrict__ A, int ldA, long long sA,
    const TB* __restrict__ B, int ldB, long long sB,
    TC* __restrict__ C, int ldC, long long sC,
    const float* __restrict__ bias, float scale, int K)
{
  __shared__ u16 As[2][BM * BKP];
  __shared__ u16 Bs[2][BN * BKP];
  const int tid  = threadIdx.x;
  const int lane = tid & 31;
  const int wave = tid >> 5;
  const int wm = wave & 3;   // wave M tile (32 rows)
  const int wn = wave >> 2;  // wave N tile (32 cols)
  const int m0 = blockIdx.y * BM;
  const int n0 = blockIdx.x * BN;
  A += (size_t)blockIdx.z * sA;
  B += (size_t)blockIdx.z * sB;
  C += (size_t)blockIdx.z * sC;

  // staging maps
  const int ar  = tid >> 1;        // 0..127 A tile row
  const int ak  = (tid & 1) << 4;  // 0,16   (stages 16 elems = two 8-chunks)
  const int br  = tid >> 2;        // 0..63  B tile row (BT path)
  const int bk2 = (tid & 3) << 3;  // 0,8,16,24
  const int bk  = tid >> 3;        // 0..31  (non-transposed B staging)
  const int bn  = (tid & 7) << 3;  // 0..56

  v8f acc[2][2] = {};

#define STAGE_TILES(to, kk)                                                       \
  {                                                                               \
    const TA* ga = A + (size_t)(m0 + ar) * ldA + (kk) + ak;                       \
    stage8v(ga,     &As[to][ar * BKP + ak]);                                      \
    stage8v(ga + 8, &As[to][ar * BKP + ak + 8]);                                  \
    if constexpr (BT) {                                                           \
      stage8v(B + (size_t)(n0 + br) * ldB + (kk) + bk2, &Bs[to][br * BKP + bk2]); \
    } else {                                                                      \
      float v[8];                                                                 \
      read8f(B + (size_t)((kk) + bk) * ldB + n0 + bn, v);                         \
      _Pragma("unroll")                                                           \
      for (int j = 0; j < 8; ++j) Bs[to][(bn + j) * BKP + bk] = f2bf(v[j]);       \
    }                                                                             \
  }

  const int nk = K / BK;
  STAGE_TILES(0, 0);
  async_wait_then_sync();

  for (int kt = 0; kt < nk; ++kt) {
    const int cur = kt & 1;
    if (kt + 1 < nk) STAGE_TILES(cur ^ 1, (kt + 1) * BK);

    v16bf a0 = frag_a(&As[cur][(wm * 32) * BKP], lane);
    v16bf a1 = frag_a(&As[cur][(wm * 32 + 16) * BKP], lane);
    v16bf b0 = frag_b(&Bs[cur][(wn * 32) * BKP], lane);
    v16bf b1 = frag_b(&Bs[cur][(wn * 32 + 16) * BKP], lane);
    acc[0][0] = __builtin_amdgcn_wmma_f32_16x16x32_bf16(false, a0, false, b0, (short)0, acc[0][0], false, false);
    acc[0][1] = __builtin_amdgcn_wmma_f32_16x16x32_bf16(false, a0, false, b1, (short)0, acc[0][1], false, false);
    acc[1][0] = __builtin_amdgcn_wmma_f32_16x16x32_bf16(false, a1, false, b0, (short)0, acc[1][0], false, false);
    acc[1][1] = __builtin_amdgcn_wmma_f32_16x16x32_bf16(false, a1, false, b1, (short)0, acc[1][1], false, false);

    async_wait_then_sync();
  }
#undef STAGE_TILES

  // C/D layout: lane<16 -> N=lane, rows r ; lane>=16 -> N=lane-16, rows 8+r
  const int lrow = (lane >> 4) << 3;
  const int lcol = lane & 15;
#pragma unroll
  for (int tm = 0; tm < 2; ++tm) {
#pragma unroll
    for (int tn = 0; tn < 2; ++tn) {
      const v8f a = acc[tm][tn];
      const int row = m0 + wm * 32 + tm * 16 + lrow;
      const int col = n0 + wn * 32 + tn * 16 + lcol;
      const float bb = bias ? bias[col] : 0.0f;
#pragma unroll
      for (int r = 0; r < 8; ++r) {
        float v = a[r] * scale + bb;
        if (ACT == 1) v = fmaxf(v, 0.0f);
        store_c(&C[(size_t)(row + r) * ldC + col], v);
      }
    }
  }
}

// rel_key[b][t][s][h] = tanh(s_key[b][s][h] + t_key[b][t][h]) -> bf16, 4 elems/thread
__global__ __launch_bounds__(256) void relkey_tanh(const float* __restrict__ sk,
                                                   const float* __restrict__ tk,
                                                   u16* __restrict__ rel) {
  const size_t i = ((size_t)blockIdx.x * 256 + threadIdx.x) * 4;
  const int h = (int)(i % 768);
  size_t rest = i / 768;
  const int s = (int)(rest % 64); rest /= 64;
  const int t = (int)(rest % 64);
  const int b = (int)(rest / 64);
  const float4 sv = *(const float4*)(sk + ((size_t)b * 64 + s) * 768 + h);
  const float4 tv = *(const float4*)(tk + ((size_t)b * 64 + t) * 768 + h);
  u16x4 o;
  o[0] = f2bf(fast_tanh(sv.x + tv.x));
  o[1] = f2bf(fast_tanh(sv.y + tv.y));
  o[2] = f2bf(fast_tanh(sv.z + tv.z));
  o[3] = f2bf(fast_tanh(sv.w + tv.w));
  *(u16x4*)(rel + i) = o;
}

// in-place softmax over rows of 4096 bf16; one block per row
__global__ __launch_bounds__(256) void softmax_rows(u16* __restrict__ sc) {
  __shared__ float red[256];
  const int tid = threadIdx.x;
  u16* p = sc + (size_t)blockIdx.x * 4096;
  float v[16];
  float m = -1e30f;
#pragma unroll
  for (int i = 0; i < 16; ++i) { v[i] = bf2f(p[tid + (i << 8)]); m = fmaxf(m, v[i]); }
  red[tid] = m; __syncthreads();
  for (int o = 128; o > 0; o >>= 1) { if (tid < o) red[tid] = fmaxf(red[tid], red[tid + o]); __syncthreads(); }
  m = red[0]; __syncthreads();
  float s = 0.0f;
#pragma unroll
  for (int i = 0; i < 16; ++i) { v[i] = __expf(v[i] - m); s += v[i]; }
  red[tid] = s; __syncthreads();
  for (int o = 128; o > 0; o >>= 1) { if (tid < o) red[tid] += red[tid + o]; __syncthreads(); }
  const float inv = 1.0f / red[0];
#pragma unroll
  for (int i = 0; i < 16; ++i) p[tid + (i << 8)] = f2bf(v[i] * inv);
}

// x[row][0:768) = bf16(query), x[row][768:1536) = ctx ; 4 elems/thread
__global__ __launch_bounds__(256) void concat_x(const float* __restrict__ query,
                                                const u16* __restrict__ ctx,
                                                u16* __restrict__ x) {
  const size_t i = ((size_t)blockIdx.x * 256 + threadIdx.x) * 4;
  const int col = (int)(i % 1536);
  const size_t row = i / 1536;
  u16x4 o;
  if (col < 768) {
    const float4 q = *(const float4*)(query + row * 768 + col);
    o[0] = f2bf(q.x); o[1] = f2bf(q.y); o[2] = f2bf(q.z); o[3] = f2bf(q.w);
  } else {
    o = *(const u16x4*)(ctx + row * 768 + (col - 768));
  }
  *(u16x4*)(x + i) = o;
}

extern "C" void kernel_launch(void* const* d_in, const int* in_sizes, int n_in,
                              void* d_out, int out_size, void* d_ws, size_t ws_size,
                              hipStream_t stream) {
  const float* query = (const float*)d_in[0];   // (32,256,768)
  const float* src   = (const float*)d_in[1];   // (32,64,1024)
  const float* trg   = (const float*)d_in[2];   // (32,64,1024)
  const float* Wq = (const float*)d_in[3]; const float* bq = (const float*)d_in[4];
  const float* Ws = (const float*)d_in[5]; const float* bs = (const float*)d_in[6];
  const float* Wt = (const float*)d_in[7]; const float* bt = (const float*)d_in[8];
  const float* Wo = (const float*)d_in[9]; const float* bo = (const float*)d_in[10];
  float* out = (float*)d_out;                   // (32,256,768) fp32

  // workspace layout (all offsets 256B-aligned)
  char* w = (char*)d_ws;
  u16*   q_bf   = (u16*)w;  w += (size_t)8192 * 768 * 2;        // q bf16
  float* s_key  = (float*)w; w += (size_t)2048 * 768 * 4;       // s_key fp32
  float* t_key  = (float*)w; w += (size_t)2048 * 768 * 4;       // t_key fp32
  u16*   rel    = (u16*)w;  w += (size_t)32 * 4096 * 768 * 2;   // rel_key bf16
  u16*   scores = (u16*)w;  w += (size_t)32 * 256 * 4096 * 2;   // scores/probs bf16
  u16*   ctx    = (u16*)w;  w += (size_t)8192 * 768 * 2;        // rel_ctx bf16
  u16*   xcat   = (u16*)w;  w += (size_t)8192 * 1536 * 2;       // concat bf16

  const float rs = 0.03608439182435161f;  // 1/sqrt(768)

  // 1) q = query @ Wq + bq            (8192x768, K=768) -> bf16
  gemm_bias_act<float, float, u16, false, 0><<<dim3(12, 64, 1), 256, 0, stream>>>(
      query, 768, 0, Wq, 768, 0, q_bf, 768, 0, bq, 1.0f, 768);
  // 2) s_key = src @ Ws + bs          (2048x768, K=1024) -> fp32
  gemm_bias_act<float, float, float, false, 0><<<dim3(12, 16, 1), 256, 0, stream>>>(
      src, 1024, 0, Ws, 768, 0, s_key, 768, 0, bs, 1.0f, 1024);
  // 3) t_key = trg @ Wt + bt
  gemm_bias_act<float, float, float, false, 0><<<dim3(12, 16, 1), 256, 0, stream>>>(
      trg, 1024, 0, Wt, 768, 0, t_key, 768, 0, bt, 1.0f, 1024);
  // 4) rel_key = tanh(s_key + t_key)  (32*4096*768 elems / 4 per thread)
  relkey_tanh<<<98304, 256, 0, stream>>>(s_key, t_key, rel);
  // 5) scores = (q @ rel_key^T) / sqrt(H), batched (M=256, N=4096, K=768); B pre-transposed
  gemm_bias_act<u16, u16, u16, true, 0><<<dim3(64, 2, 32), 256, 0, stream>>>(
      q_bf, 768, 256LL * 768, rel, 768, 4096LL * 768,
      scores, 4096, 256LL * 4096, nullptr, rs, 768);
  // 6) softmax over ts=4096, in place
  softmax_rows<<<8192, 256, 0, stream>>>(scores);
  // 7) rel_ctx = probs @ rel_key, batched (M=256, N=768, K=4096)
  gemm_bias_act<u16, u16, u16, false, 0><<<dim3(12, 2, 32), 256, 0, stream>>>(
      scores, 4096, 256LL * 4096, rel, 768, 4096LL * 768,
      ctx, 768, 256LL * 768, nullptr, 1.0f, 4096);
  // 8) x = concat(query, rel_ctx) -> bf16 (8192 x 1536)
  concat_x<<<12288, 256, 0, stream>>>(query, ctx, xcat);
  // 9) out = relu(x @ Wo + bo)        (8192x768, K=1536) -> fp32
  gemm_bias_act<u16, float, float, false, 1><<<dim3(12, 64, 1), 256, 0, stream>>>(
      xcat, 1536, 0, Wo, 768, 0, out, 768, 0, bo, 1.0f, 1536);
}
